// PraxisCompressiveMemory_70901320122577
// MI455X (gfx1250) — matrix operand order repaired and checked by reading, not compiled, not run on hardware
//
#include <hip/hip_runtime.h>
#include <hip/hip_bf16.h>

// ---------------------------------------------------------------------------
// Infini-Attention compressive memory (delta rule), single segment.
// B=4, H=32, S=2048, D=128.  f32 I/O, f16 WMMA (f32 accumulate) for matmuls.
// Async (ASYNCcnt) LDS staging for the memory matrix when available.
// ---------------------------------------------------------------------------

#define BB 4
#define HH 32
#define SS 2048
#define DD 128
#define STILE 128          // s-rows per workgroup
#define LDST 136           // LDS row stride in halfs (pad to break bank conflicts)

typedef __attribute__((ext_vector_type(16))) _Float16 v16h;
typedef __attribute__((ext_vector_type(8)))  _Float16 v8h;
typedef __attribute__((ext_vector_type(8)))  float    v8f;
typedef __attribute__((ext_vector_type(4)))  int      v4i;

#if defined(__AMDGCN__) && __has_builtin(__builtin_amdgcn_global_load_async_to_lds_b128) && __has_builtin(__builtin_amdgcn_s_wait_asynccnt)
#define USE_ASYNC_LDS 1
typedef __attribute__((address_space(1))) v4i glb_v4i;
typedef __attribute__((address_space(3))) v4i lds_v4i;
#else
#define USE_ASYNC_LDS 0
#endif

// LDS layout (single carved arena so the M f32 staging area can alias bufQ/bufK)
#define OFF_BUFM   0u                    // 128*136*2 = 34816 B : M^T then sigma_k^T
#define OFF_BUFQ   34816u                // 34816 B : sigma_q then delta^T
#define OFF_BUFK   69632u                // 34816 B : sigma_k (row major)
#define OFF_RDQ    104448u               // 512 B
#define OFF_RDK    104960u               // 512 B
#define OFF_ZSH    105472u               // 512 B
#define OFF_PARTQ  105984u               // 1024 B
#define OFF_PARTK  107008u               // 1024 B
#define SMEM_BYTES 108032u
#define OFF_STAGE  OFF_BUFQ              // 65536 B f32 staging aliases bufQ+bufK

// ---- WMMA operand loaders -------------------------------------------------
// A operand (16x32 f16): lane L(0..15) = row, VGPR0-3 hold K = kk*32 + lh*8 .. +7,
// VGPR4-7 hold K = kk*32 + lh*8 + 16 .. +23  (lh = lane>>4).
__device__ __forceinline__ v16h lds_a(const _Float16* base, int row, int kk, int lane) {
  const int off = row * LDST + kk * 32 + ((lane >> 4) << 3);
  v8h lo = *(const v8h*)(base + off);
  v8h hi = *(const v8h*)(base + off + 16);
  v16h a;
#pragma unroll
  for (int i = 0; i < 8; ++i) { a[i] = lo[i]; a[i + 8] = hi[i]; }
  return a;
}
// B operand (32x16 f16) from COLUMN-MAJOR storage (row of 'base' = output col):
// lane L(0..15) = col, VGPR0-3 hold K = kk*32 + lh*8 .. +7, VGPR4-7 hold K = kk*32 + lh*8 + 16 .. +23.
__device__ __forceinline__ v16h lds_b(const _Float16* base, int col, int kk, int lane) {
  const int off = col * LDST + kk * 32 + ((lane >> 4) << 3);
  v8h lo = *(const v8h*)(base + off);
  v8h hi = *(const v8h*)(base + off + 16);
  v16h b;
#pragma unroll
  for (int i = 0; i < 8; ++i) { b[i] = lo[i]; b[i + 8] = hi[i]; }
  return b;
}

// ---- init: new_states = init_mem (bcast over B), new_z = init_z -----------
__global__ __launch_bounds__(256) void init_states_kernel(
    const float* __restrict__ mem0, const float* __restrict__ z0,
    float* __restrict__ st, float* __restrict__ zo) {
  const size_t i = (size_t)blockIdx.x * 256 + threadIdx.x;
  const size_t nst = (size_t)BB * HH * DD * DD;
  if (i < nst) {
    const int h = (int)((i / (DD * DD)) % HH);
    st[i] = mem0[(size_t)h * DD * DD + (i % (DD * DD))];
  }
  const size_t nz = (size_t)BB * HH * DD;
  if (i < nz) {
    const int h = (int)((i / DD) % HH);
    zo[i] = z0[(size_t)h * DD + (i % DD)];
  }
}

// ---- main kernel ----------------------------------------------------------
__global__ __launch_bounds__(256) void infini_main_kernel(
    const float* __restrict__ q, const float* __restrict__ k,
    const float* __restrict__ v, const float* __restrict__ attn,
    const float* __restrict__ betas, const float* __restrict__ mem0,
    const float* __restrict__ z0,
    float* __restrict__ out, float* __restrict__ st, float* __restrict__ zo) {
  __shared__ __align__(16) unsigned char smem[SMEM_BYTES];
  _Float16* bufM = (_Float16*)(smem + OFF_BUFM);
  _Float16* bufQ = (_Float16*)(smem + OFF_BUFQ);
  _Float16* bufK = (_Float16*)(smem + OFF_BUFK);
  float* rdq   = (float*)(smem + OFF_RDQ);
  float* rdk   = (float*)(smem + OFF_RDK);
  float* zsh   = (float*)(smem + OFF_ZSH);
  float* partq = (float*)(smem + OFF_PARTQ);
  float* partk = (float*)(smem + OFF_PARTK);

  const int tid = threadIdx.x;
  const int ntile = SS / STILE;
  const int tileS = blockIdx.x % ntile;
  const int bh = blockIdx.x / ntile;
  const int h = bh % HH;
  const int s0 = tileS * STILE;
  const size_t baseG = ((size_t)bh * SS + s0) * DD;  // into q/k/v/attn/out
  const float* memH = mem0 + (size_t)h * DD * DD;
  const float* zH = z0 + (size_t)h * DD;

  // ---- P0: stage z and M^T (f32 -> f16) ----
  if (tid < DD) zsh[tid] = zH[tid];
#if USE_ASYNC_LDS
  {
    float* stage = (float*)(smem + OFF_STAGE);     // aliases bufQ/bufK (dead here)
#pragma unroll
    for (int i = 0; i < 16; ++i) {
      const int idx = (tid + i * 256) * 4;         // 16 x 256 x 4 floats = 16384
      __builtin_amdgcn_global_load_async_to_lds_b128(
          (glb_v4i*)(memH + idx), (lds_v4i*)(stage + idx), 0, 0);
    }
    __builtin_amdgcn_s_wait_asynccnt(0);
    __syncthreads();                               // all waves' async writes visible
#pragma unroll
    for (int i = 0; i < 64; ++i) {
      const int idx = tid + i * 256;
      bufM[(idx & 127) * LDST + (idx >> 7)] = (_Float16)stage[idx];
    }
  }
#else
#pragma unroll
  for (int i = 0; i < 64; ++i) {
    const int idx = tid + i * 256;                 // coalesced global read
    bufM[(idx & 127) * LDST + (idx >> 7)] = (_Float16)memH[idx];
  }
#endif
  __syncthreads();

  // ---- P1: sigma_q / sigma_k + dots with z ----
  {
    const int s = tid >> 1;
    const int c0 = (tid & 1) * 64;
    const float* qrow = q + baseG + (size_t)s * DD + c0;
    const float* krow = k + baseG + (size_t)s * DD + c0;
    float dq = 0.f, dk = 0.f;
#pragma unroll 8
    for (int c = 0; c < 64; ++c) {
      float x = qrow[c];
      float sq = x > 0.f ? x + 1.f : __expf(x);    // elu(x)+1
      bufQ[s * LDST + c0 + c] = (_Float16)sq;
      dq += sq * zsh[c0 + c];
      float y = krow[c];
      float sk = y > 0.f ? y + 1.f : __expf(y);
      bufK[s * LDST + c0 + c] = (_Float16)sk;
      dk += sk * zsh[c0 + c];
    }
    partq[tid] = dq;
    partk[tid] = dk;
  }
  __syncthreads();
  if (tid < STILE) {
    // store RECIPROCALS: one v_rcp_f32 per row instead of per-element IEEE divides
    rdq[tid] = __builtin_amdgcn_rcpf(partq[2 * tid] + partq[2 * tid + 1]);
    rdk[tid] = __builtin_amdgcn_rcpf(partk[2 * tid] + partk[2 * tid + 1]);
  }
  __syncthreads();

  const int lane = tid & 31;
  const int wv = tid >> 5;        // wave id 0..7
  const int m0 = wv * 16;         // 16 output rows per wave
  const int arow = m0 + (lane & 15);
  const int lcol = lane & 15;
  const int lhi = lane >> 4;

  // ---- P2: memory_output = (sigma_q @ M) * recip(sigma_q . z); gated blend ----
  for (int nt = 0; nt < 8; ++nt) {
    v8f acc = {};
#pragma unroll
    for (int kk = 0; kk < 4; ++kk) {
      v16h a = lds_a(bufQ, arow, kk, lane);
      v16h bm = lds_b(bufM, nt * 16 + lcol, kk, lane);
      acc = __builtin_amdgcn_wmma_f32_16x16x32_f16(false, a, false, bm,
                                                   (short)0, acc, false, false);
    }
    const int col = nt * 16 + lcol;
    const float g = __builtin_amdgcn_rcpf(1.f + __expf(-betas[h * DD + col]));
#pragma unroll
    for (int r = 0; r < 8; ++r) {
      const int m = m0 + r + (lhi << 3);
      const float mo = acc[r] * rdq[m];
      const size_t gi = baseG + (size_t)m * DD + col;
      out[gi] = g * mo + (1.f - g) * attn[gi];
    }
  }
  __syncthreads();  // bufQ (sigma_q) dead; becomes delta^T

  // ---- P3: retrieved = (sigma_k @ M) * recip(sigma_k . z); delta = v - retrieved ----
  for (int nt = 0; nt < 8; ++nt) {
    v8f acc = {};
#pragma unroll
    for (int kk = 0; kk < 4; ++kk) {
      v16h a = lds_a(bufK, arow, kk, lane);
      v16h bm = lds_b(bufM, nt * 16 + lcol, kk, lane);
      acc = __builtin_amdgcn_wmma_f32_16x16x32_f16(false, a, false, bm,
                                                   (short)0, acc, false, false);
    }
    const int col = nt * 16 + lcol;
#pragma unroll
    for (int r = 0; r < 8; ++r) {
      const int m = m0 + r + (lhi << 3);
      const float ret = acc[r] * rdk[m];
      const float dv = v[baseG + (size_t)m * DD + col] - ret;
      bufQ[col * LDST + m] = (_Float16)dv;   // delta^T: [col][s]
    }
  }
  __syncthreads();  // bufM (M^T) dead; becomes sigma_k^T

  // ---- P3.5: transpose sigma_k into bufM ----
  {
    const int s = tid >> 1;
    const int c0 = (tid & 1) * 64;
#pragma unroll 8
    for (int c = 0; c < 64; ++c)
      bufM[(c0 + c) * LDST + s] = bufK[s * LDST + c0 + c];
  }
  __syncthreads();

  // ---- P4: new_states += sigma_k^T @ delta  (atomic f32 accumulate) ----
  for (int ct = 0; ct < 8; ++ct) {
    v8f acc = {};
#pragma unroll
    for (int kk = 0; kk < 4; ++kk) {
      v16h a = lds_a(bufM, arow, kk, lane);              // rows = d
      v16h bb = lds_b(bufQ, ct * 16 + lcol, kk, lane);   // delta^T cols
      acc = __builtin_amdgcn_wmma_f32_16x16x32_f16(false, a, false, bb,
                                                   (short)0, acc, false, false);
    }
#pragma unroll
    for (int r = 0; r < 8; ++r) {
      const int drow = m0 + r + (lhi << 3);
      const int dcol = ct * 16 + lcol;
      unsafeAtomicAdd(&st[((size_t)bh * DD + drow) * DD + dcol], acc[r]);
    }
  }

  // ---- P5: new_z += column sums of sigma_k ----
  if (tid < DD) {
    float sum = 0.f;
    for (int s = 0; s < STILE; ++s) sum += (float)bufM[tid * LDST + s];
    unsafeAtomicAdd(&zo[(size_t)bh * DD + tid], sum);
  }
}

// ---------------------------------------------------------------------------
extern "C" void kernel_launch(void* const* d_in, const int* in_sizes, int n_in,
                              void* d_out, int out_size, void* d_ws, size_t ws_size,
                              hipStream_t stream) {
  const float* q     = (const float*)d_in[0];
  const float* k     = (const float*)d_in[1];
  const float* v     = (const float*)d_in[2];
  const float* attn  = (const float*)d_in[3];
  const float* betas = (const float*)d_in[4];
  const float* mem0  = (const float*)d_in[5];
  const float* z0    = (const float*)d_in[6];

  float* out = (float*)d_out;                         // [B,H,S,D]
  float* st  = out + (size_t)BB * HH * SS * DD;       // [B,H,D,D]
  float* zo  = st + (size_t)BB * HH * DD * DD;        // [B,H,D,1]

  const int nInit = (BB * HH * DD * DD + 255) / 256;
  init_states_kernel<<<nInit, 256, 0, stream>>>(mem0, z0, st, zo);

  const int nMain = BB * HH * (SS / STILE);           // 2048 workgroups
  infini_main_kernel<<<nMain, 256, 0, stream>>>(q, k, v, attn, betas, mem0, z0,
                                                out, st, zo);
}
